// ROMO_26766236188969
// MI455X (gfx1250) — matrix operand order repaired and checked by reading, not compiled, not run on hardware
//
#include <hip/hip_runtime.h>
#include <hip/hip_bf16.h>

// ---------------------------------------------------------------------------
// Types for CDNA5 WMMA (wave32, 16x16x32 bf16 -> f32)
// ---------------------------------------------------------------------------
typedef __bf16 bf16_t;
typedef bf16_t v16bf __attribute__((ext_vector_type(16)));
typedef float  v8f   __attribute__((ext_vector_type(8)));
typedef unsigned short ushort_t;
typedef unsigned int   uint32_tt;

union V16 { uint4 q[2]; v16bf v; };

__device__ __forceinline__ ushort_t f2bf(float x) {
    uint32_tt u = __float_as_uint(x);
    u = (u + 0x7FFFu + ((u >> 16) & 1u)) >> 16;   // round-to-nearest-even
    return (ushort_t)u;
}
__device__ __forceinline__ float bf2f(ushort_t h) {
    return __uint_as_float(((uint32_tt)h) << 16);
}

// ---------------------------------------------------------------------------
// Generic bf16 WMMA GEMM:  C = act(A(MxK) @ B(KxN) + bias)
//   A, B bf16 row-major.  Block tile 128x128, K chunks of 32 staged in LDS
//   (B transposed in LDS so each lane's operand is contiguous).
//   8 waves, each wave owns a 32x64 sub-tile = 2x4 accumulators.
// ---------------------------------------------------------------------------
#define GTILE 128
#define GKC   32
#define ASTR  40   // LDS row stride (elements), 80B -> 16B aligned
#define BSTR  40

template <int RELU, int OUTBF>
__global__ __launch_bounds__(256) void k_gemm(
    const ushort_t* __restrict__ A, int lda,
    const ushort_t* __restrict__ B, int ldb,
    const float*   __restrict__ bias,
    void*          __restrict__ C, int ldc,
    int N, int K, int nLimit)
{
    __shared__ ushort_t As[GTILE * ASTR];
    __shared__ ushort_t Bt[GTILE * BSTR];

    const int tid  = threadIdx.x;
    const int lane = tid & 31;
    const int w    = tid >> 5;
    const int wr   = w >> 1;     // 0..3 : 32-row band
    const int wc   = w & 1;      // 0..1 : 64-col band
    const int rBase = blockIdx.y * GTILE;
    const int nBase = blockIdx.x * GTILE;

    v8f acc[2][4];
#pragma unroll
    for (int i = 0; i < 2; ++i)
#pragma unroll
        for (int j = 0; j < 4; ++j)
#pragma unroll
            for (int e = 0; e < 8; ++e) acc[i][j][e] = 0.0f;

    const int nChunks = K / GKC;
    for (int kc = 0; kc < nChunks; ++kc) {
        __syncthreads();
        {   // stage A chunk: 128 rows x 32 k, 16 bf16 per thread
            int r  = tid >> 1;
            int kk = (tid & 1) * 16;
            const uint4* src = (const uint4*)(A + (size_t)(rBase + r) * lda + kc * GKC + kk);
            uint4* dst = (uint4*)(As + r * ASTR + kk);
            dst[0] = src[0];
            dst[1] = src[1];
        }
        {   // stage B chunk transposed: Bt[n][k]
            int kk = tid & 31;
            int nb = (tid >> 5) * 16;
            int gk = kc * GKC + kk;
#pragma unroll
            for (int i = 0; i < 16; ++i) {
                int col = nBase + nb + i;
                ushort_t v = (col < N) ? B[(size_t)gk * ldb + col] : (ushort_t)0;
                Bt[(nb + i) * BSTR + kk] = v;
            }
        }
        __syncthreads();

        // A operands: lane holds row M=lane&15 (+wave/tile base);
        // elements 0..7 -> K=kb..kb+7, 8..15 -> K=kb+16..kb+23, kb=8*(lane>=16)
        const int arow0 = wr * 32 + (lane & 15);
        const int kbA   = (lane & 16) ? 8 : 0;
        V16 a[2];
#pragma unroll
        for (int mt = 0; mt < 2; ++mt) {
            const ushort_t* p = As + (arow0 + mt * 16) * ASTR + kbA;
            a[mt].q[0] = *(const uint4*)(p);
            a[mt].q[1] = *(const uint4*)(p + 16);
        }
        // B operands: lane holds col N=lane&15; elements 0..7 -> K=kb..kb+7,
        // 8..15 -> K=kb+16..kb+23, kb=8*(lane>=16). Bt is column-major(-transposed).
        const int bcol0 = wc * 64 + (lane & 15);
        const int kbB   = (lane & 16) ? 8 : 0;
        V16 b[4];
#pragma unroll
        for (int nt = 0; nt < 4; ++nt) {
            const ushort_t* p = Bt + (bcol0 + nt * 16) * BSTR + kbB;
            b[nt].q[0] = *(const uint4*)(p);
            b[nt].q[1] = *(const uint4*)(p + 16);
        }
#pragma unroll
        for (int mt = 0; mt < 2; ++mt)
#pragma unroll
            for (int nt = 0; nt < 4; ++nt)
                acc[mt][nt] = __builtin_amdgcn_wmma_f32_16x16x32_bf16(
                    false, a[mt].v, false, b[nt].v, (short)0, acc[mt][nt], false, false);
    }

    // Epilogue: C/D layout: VGPR d, lanes 0-15 -> M=d, lanes 16-31 -> M=8+d; N=lane&15
    const int mlo = (lane >> 4) * 8;
    const int nl  = lane & 15;
#pragma unroll
    for (int mt = 0; mt < 2; ++mt) {
#pragma unroll
        for (int nt = 0; nt < 4; ++nt) {
            int col = nBase + wc * 64 + nt * 16 + nl;
            if (col >= nLimit) continue;
            float bv = bias ? bias[col] : 0.0f;
#pragma unroll
            for (int d = 0; d < 8; ++d) {
                int row = rBase + wr * 32 + mt * 16 + mlo + d;
                float v = acc[mt][nt][d] + bv;
                if (RELU) v = fmaxf(v, 0.0f);
                if (OUTBF) ((ushort_t*)C)[(size_t)row * ldc + col] = f2bf(v);
                else       ((float*)C)[(size_t)row * ldc + col]    = v;
            }
        }
    }
}

// ---------------------------------------------------------------------------
// Fused per-batch kernel: k-projection (WMMA) + distances + softmax + value agg
// One block per batch item b. retrieval_set[b] staged once into LDS as bf16.
//   vagg[b, 0:129] = sum_r softmax(1/max(||q_b - k_r||,1e-8))_r * value[b,r,:]
//   (cols 129..159 zero-padded for the following WMMA GEMM)
// ---------------------------------------------------------------------------
#define RSTR 136   // LDS row stride for 129-wide rows (272B, 16B aligned)

__global__ __launch_bounds__(256) void k_dist(
    const float*    __restrict__ q,     // 4096 x 128 (f32, already projected)
    const float*    __restrict__ rs,    // 4096 x 128 x 129
    const ushort_t* __restrict__ Wk,    // 128 x 128 bf16
    const float*    __restrict__ bk,    // 128
    ushort_t*       __restrict__ vagg)  // 4096 x 160 bf16 (zero padded)
{
    __shared__ ushort_t rsb[128 * RSTR];
    __shared__ ushort_t Bt[128 * BSTR];
    __shared__ float qs[128];
    __shared__ float d2[128];
    __shared__ float warr[128];

    const int b    = blockIdx.x;
    const int tid  = threadIdx.x;
    const int lane = tid & 31;
    const int w    = tid >> 5;
    const int wr   = w >> 1;
    const int wc   = w & 1;

    if (tid < 128) { qs[tid] = q[(size_t)b * 128 + tid]; d2[tid] = 0.0f; }

    const float* rsrc = rs + (size_t)b * 128 * 129;
    for (int idx = tid; idx < 128 * 129; idx += 256) {
        int r = idx / 129;
        int c = idx - 129 * r;
        rsb[r * RSTR + c] = f2bf(rsrc[idx]);
    }

    v8f acc[2][4];
#pragma unroll
    for (int i = 0; i < 2; ++i)
#pragma unroll
        for (int j = 0; j < 4; ++j)
#pragma unroll
            for (int e = 0; e < 8; ++e) acc[i][j][e] = 0.0f;

    // Kp = key(128x128) @ Wk(128x128), K chunks of 32
    for (int kc = 0; kc < 4; ++kc) {
        __syncthreads();
        {
            int kk = tid & 31;
            int nb = (tid >> 5) * 16;
            int gk = kc * 32 + kk;
#pragma unroll
            for (int i = 0; i < 16; ++i)
                Bt[(nb + i) * BSTR + kk] = Wk[(size_t)gk * 128 + nb + i];
        }
        __syncthreads();

        const int arow0 = wr * 32 + (lane & 15);
        const int kbA   = (lane & 16) ? 8 : 0;
        V16 a[2];
#pragma unroll
        for (int mt = 0; mt < 2; ++mt) {
            const ushort_t* p = rsb + (arow0 + mt * 16) * RSTR + kc * 32 + kbA;
            a[mt].q[0] = *(const uint4*)(p);
            a[mt].q[1] = *(const uint4*)(p + 16);
        }
        const int bcol0 = wc * 64 + (lane & 15);
        const int kbB   = (lane & 16) ? 8 : 0;
        V16 bb[4];
#pragma unroll
        for (int nt = 0; nt < 4; ++nt) {
            const ushort_t* p = Bt + (bcol0 + nt * 16) * BSTR + kbB;
            bb[nt].q[0] = *(const uint4*)(p);
            bb[nt].q[1] = *(const uint4*)(p + 16);
        }
#pragma unroll
        for (int mt = 0; mt < 2; ++mt)
#pragma unroll
            for (int nt = 0; nt < 4; ++nt)
                acc[mt][nt] = __builtin_amdgcn_wmma_f32_16x16x32_bf16(
                    false, a[mt].v, false, bb[nt].v, (short)0, acc[mt][nt], false, false);
    }

    // squared distances from accumulators; row r fixed per (mt,d,lane-half)
    const int mlo = (lane >> 4) * 8;
    const int nl  = lane & 15;
    float part[2][8];
#pragma unroll
    for (int mt = 0; mt < 2; ++mt)
#pragma unroll
        for (int d = 0; d < 8; ++d) part[mt][d] = 0.0f;

#pragma unroll
    for (int mt = 0; mt < 2; ++mt)
#pragma unroll
        for (int nt = 0; nt < 4; ++nt) {
            int j = wc * 64 + nt * 16 + nl;
            float bkj = bk[j], qj = qs[j];
#pragma unroll
            for (int d = 0; d < 8; ++d) {
                float kp   = acc[mt][nt][d] + bkj;
                float diff = qj - kp;
                part[mt][d] += diff * diff;
            }
        }
#pragma unroll
    for (int mt = 0; mt < 2; ++mt)
#pragma unroll
        for (int d = 0; d < 8; ++d) {
            float v = part[mt][d];
            v += __shfl_xor(v, 1, 16);
            v += __shfl_xor(v, 2, 16);
            v += __shfl_xor(v, 4, 16);
            v += __shfl_xor(v, 8, 16);
            if (nl == 0) {
                int r = wr * 32 + mt * 16 + mlo + d;
                atomicAdd(&d2[r], v);   // ds_add_f32; two n-band waves per r
            }
        }
    __syncthreads();

    // softmax over 1/dist
    if (tid < 128) {
        float dd = fmaxf(sqrtf(d2[tid]), 1e-8f);
        warr[tid] = 1.0f / dd;
    }
    __syncthreads();
    float e = 0.0f;
    if (tid < 128) {
        float m = -3.4e38f;
        for (int i = 0; i < 128; ++i) m = fmaxf(m, warr[i]);
        e = __expf(warr[tid] - m);
    }
    __syncthreads();
    if (tid < 128) d2[tid] = e;
    __syncthreads();
    if (tid < 128) {
        float s = 0.0f;
        for (int i = 0; i < 128; ++i) s += d2[i];
        warr[tid] = e / s;
    }
    __syncthreads();

    // value aggregation (values aggregated BEFORE Wv projection: exact by linearity)
    for (int j = tid; j < 160; j += 256) {
        float s = 0.0f;
        if (j < 129) {
            for (int r = 0; r < 128; ++r) s += warr[r] * bf2f(rsb[r * RSTR + j]);
        }
        vagg[(size_t)b * 160 + j] = f2bf(s);
    }
}

// ---------------------------------------------------------------------------
// Final N=1 layer: out[row] = dot(H_bf16[row,:], W_f32) + b   (wave per row)
// ---------------------------------------------------------------------------
__global__ __launch_bounds__(256) void k_dot(
    const ushort_t* __restrict__ H, const float* __restrict__ W,
    const float* __restrict__ bsc, float* __restrict__ out, int K)
{
    int w    = threadIdx.x >> 5;
    int lane = threadIdx.x & 31;
    int row  = blockIdx.x * 8 + w;
    const ushort_t* hp = H + (size_t)row * K;
    float acc = 0.0f;
    for (int k = lane; k < K; k += 32) acc += bf2f(hp[k]) * W[k];
    for (int m = 16; m > 0; m >>= 1) acc += __shfl_xor(acc, m, 32);
    if (lane == 0) out[row] = acc + bsc[0];
}

// ---------------------------------------------------------------------------
// Elementwise helpers
// ---------------------------------------------------------------------------
__global__ void k_conv_pad(const float* __restrict__ src, ushort_t* __restrict__ dst,
                           int sR, int sC, int dR, int dC)
{
    int idx = blockIdx.x * 256 + threadIdx.x;
    int tot = dR * dC;
    if (idx >= tot) return;
    int r = idx / dC, c = idx - r * dC;
    float v = (r < sR && c < sC) ? src[(size_t)r * sC + c] : 0.0f;
    dst[idx] = f2bf(v);
}

// xcat: cols 0..127 = bf16(target), cols 257..287 = 0, cols 128..256 filled by rl GEMM
__global__ void k_build_xcat(const float* __restrict__ target, ushort_t* __restrict__ xcat)
{
    int idx = blockIdx.x * 256 + threadIdx.x;
    if (idx >= 4096 * 288) return;
    int r = idx / 288, c = idx - r * 288;
    if (c < 128)       xcat[idx] = f2bf(target[(size_t)r * 128 + c]);
    else if (c >= 257) xcat[idx] = 0;
}

__global__ void k_combine(const float* __restrict__ o0, const float* __restrict__ o1,
                          float* __restrict__ out)
{
    int i = blockIdx.x * 256 + threadIdx.x;
    if (i >= 4096) return;
    float a = o0[i], b = o1[i];
    out[i]            = 0.5f * (a + b);
    out[4096 + i]     = a;
    out[2 * 4096 + i] = b;
}

// ---------------------------------------------------------------------------
// Host launcher
// ---------------------------------------------------------------------------
static inline size_t alignUp(size_t x) { return (x + 255) & ~(size_t)255; }

extern "C" void kernel_launch(void* const* d_in, const int* in_sizes, int n_in,
                              void* d_out, int out_size, void* d_ws, size_t ws_size,
                              hipStream_t stream)
{
    (void)in_sizes; (void)n_in; (void)out_size; (void)ws_size;
    const float* target = (const float*)d_in[0];   // 4096x128
    const float* rset   = (const float*)d_in[1];   // 4096x128x129
    const float* Wq = (const float*)d_in[2];  const float* bq = (const float*)d_in[3];
    const float* Wk = (const float*)d_in[4];  const float* bk = (const float*)d_in[5];
    const float* Wv = (const float*)d_in[6];  const float* bv = (const float*)d_in[7];
    const float* W0 = (const float*)d_in[8];  const float* b0 = (const float*)d_in[9];
    const float* W1 = (const float*)d_in[10]; const float* b1 = (const float*)d_in[11];
    const float* W2 = (const float*)d_in[12]; const float* b2 = (const float*)d_in[13];
    const float* M0 = (const float*)d_in[14]; const float* c0 = (const float*)d_in[15];
    const float* M1 = (const float*)d_in[16]; const float* c1 = (const float*)d_in[17];
    const float* M2 = (const float*)d_in[18]; const float* c2 = (const float*)d_in[19];
    float* out = (float*)d_out;

    // workspace carve
    char* p = (char*)d_ws;
    auto take = [&](size_t bytes) { char* r = p; p += alignUp(bytes); return r; };
    ushort_t* tbf  = (ushort_t*)take((size_t)4096 * 128 * 2);
    ushort_t* wqb  = (ushort_t*)take((size_t)128 * 128 * 2);
    ushort_t* wkb  = (ushort_t*)take((size_t)128 * 128 * 2);
    ushort_t* wvb  = (ushort_t*)take((size_t)160 * 160 * 2);   // padded 129->160
    ushort_t* w0b  = (ushort_t*)take((size_t)288 * 2048 * 2);  // padded 257->288
    ushort_t* w1b  = (ushort_t*)take((size_t)2048 * 2048 * 2);
    ushort_t* m0b  = (ushort_t*)take((size_t)128 * 2048 * 2);
    ushort_t* m1b  = (ushort_t*)take((size_t)2048 * 2048 * 2);
    float*    qf   = (float*)take((size_t)4096 * 128 * 4);
    ushort_t* vagg = (ushort_t*)take((size_t)4096 * 160 * 2);
    ushort_t* xcat = (ushort_t*)take((size_t)4096 * 288 * 2);
    ushort_t* bufA = (ushort_t*)take((size_t)4096 * 2048 * 2); // h0 then g0
    ushort_t* bufB = (ushort_t*)take((size_t)4096 * 2048 * 2); // h1 then g1
    float*    o0   = (float*)take((size_t)4096 * 4);
    float*    o1   = (float*)take((size_t)4096 * 4);

    auto cgrid = [](size_t tot) { return dim3((unsigned)((tot + 255) / 256)); };

    // 1) conversions (re-done each call: deterministic, cheap)
    k_conv_pad<<<cgrid(4096 * 128), 256, 0, stream>>>(target, tbf, 4096, 128, 4096, 128);
    k_conv_pad<<<cgrid(128 * 128), 256, 0, stream>>>(Wq, wqb, 128, 128, 128, 128);
    k_conv_pad<<<cgrid(128 * 128), 256, 0, stream>>>(Wk, wkb, 128, 128, 128, 128);
    k_conv_pad<<<cgrid(160 * 160), 256, 0, stream>>>(Wv, wvb, 129, 129, 160, 160);
    k_conv_pad<<<cgrid(288 * 2048), 256, 0, stream>>>(W0, w0b, 257, 2048, 288, 2048);
    k_conv_pad<<<cgrid(2048 * 2048), 256, 0, stream>>>(W1, w1b, 2048, 2048, 2048, 2048);
    k_conv_pad<<<cgrid(128 * 2048), 256, 0, stream>>>(M0, m0b, 128, 2048, 128, 2048);
    k_conv_pad<<<cgrid(2048 * 2048), 256, 0, stream>>>(M1, m1b, 2048, 2048, 2048, 2048);
    k_build_xcat<<<cgrid(4096 * 288), 256, 0, stream>>>(target, xcat);

    // 2) q = target @ Wq + bq  (f32 out)
    k_gemm<0, 0><<<dim3(1, 32), 256, 0, stream>>>(tbf, 128, wqb, 128, bq,
                                                  (void*)qf, 128, 128, 128, 128);
    // 3) fused distance/softmax/value-aggregation
    k_dist<<<dim3(4096), 256, 0, stream>>>(qf, rset, wkb, bk, vagg);

    // 4) rl = vagg @ Wv + bv -> xcat columns 128..256 (bf16 out, store limit 129)
    k_gemm<0, 1><<<dim3(2, 32), 256, 0, stream>>>(vagg, 160, wvb, 160, bv,
                                                  (void*)(xcat + 128), 288, 160, 160, 129);
    // 5) MLP-0
    k_gemm<1, 1><<<dim3(16, 32), 256, 0, stream>>>(xcat, 288, w0b, 2048, b0,
                                                   (void*)bufA, 2048, 2048, 288, 2048);
    k_gemm<1, 1><<<dim3(16, 32), 256, 0, stream>>>(bufA, 2048, w1b, 2048, b1,
                                                   (void*)bufB, 2048, 2048, 2048, 2048);
    k_dot<<<dim3(512), 256, 0, stream>>>(bufB, W2, b2, o0, 2048);

    // 6) MLP-1 (reuses bufA/bufB after o0 is extracted)
    k_gemm<1, 1><<<dim3(16, 32), 256, 0, stream>>>(tbf, 128, m0b, 2048, c0,
                                                   (void*)bufA, 2048, 2048, 128, 2048);
    k_gemm<1, 1><<<dim3(16, 32), 256, 0, stream>>>(bufA, 2048, m1b, 2048, c1,
                                                   (void*)bufB, 2048, 2048, 2048, 2048);
    k_dot<<<dim3(512), 256, 0, stream>>>(bufB, M2, c2, o1, 2048);

    // 7) out tuple: [out, out0, out1]
    k_combine<<<cgrid(4096), 256, 0, stream>>>(o0, o1, out);
}
